// SCSAttention_12661563588944
// MI455X (gfx1250) — matrix operand order repaired and checked by reading, not compiled, not run on hardware
//
#include <hip/hip_runtime.h>
#include <hip/hip_bf16.h>

// ---------------------------------------------------------------------------
// Types for CDNA5 WMMA (wave32, v_wmma_f32_16x16x32_bf16)
// ---------------------------------------------------------------------------
typedef __bf16 bf16;
typedef bf16  v16bf __attribute__((ext_vector_type(16)));
typedef bf16  v8bf  __attribute__((ext_vector_type(8)));
typedef float v8f   __attribute__((ext_vector_type(8)));

// Problem constants (from reference)
#define N_IMG   32          // B*F
#define CIN_    2048
#define PX      49          // 7*7
#define CMID    256
#define CH_     128
#define NTOK    1568        // N_IMG * PX, multiple of 32
#define SVAL    392         // F * PX tokens per clip
#define SPAD    416         // padded to multiple of 32
#define NBATCH  4

// ---------------------------------------------------------------------------
// Fragment loads per CDNA5 ISA 16-bit A (16x32) / B (32x16) VGPR layouts.
// A: lane L holds row M=L%16; halves 0..7 = K{base..base+7}, 8..15 = K{base+16..}
//    with base = 8*(L>>4).
// B: lane L holds column N=L%16; same K order as A.
// ---------------------------------------------------------------------------
__device__ __forceinline__ v16bf ld_afrag(const bf16* p) {
    v8bf lo = *(const v8bf*)(p);
    v8bf hi = *(const v8bf*)(p + 16);
    return __builtin_shufflevector(lo, hi, 0,1,2,3,4,5,6,7,8,9,10,11,12,13,14,15);
}
__device__ __forceinline__ v16bf ld_bfrag(const bf16* p) {
    v8bf lo = *(const v8bf*)(p);
    v8bf hi = *(const v8bf*)(p + 16);
    return __builtin_shufflevector(lo, hi, 0,1,2,3,4,5,6,7,8,9,10,11,12,13,14,15);
}

// ---------------------------------------------------------------------------
// Store one 16x16 f32 accumulator tile with mode-dependent epilogue.
// C/D layout: lane L -> column = L%16; VGPR r -> row = r + 8*(L>>4).
// Modes:
//  0: bf16 plain          Out[row*ldo + colOff + col]
//  1: Q remap             Out[(b*416 + s)*256 + colOff + col]     (row = b*392+s)
//  2: K odd-permute       flat=((w*8+f)*7+h)*256+c -> Out[(b*416+m)*256+cc]
//  3: f32 plain           Out[row*ldo + col]
//  4: V transposed + pos  Out[(b*256+col)*416 + s] (+pos_embedV)
// ---------------------------------------------------------------------------
__device__ __forceinline__ void store_acc(v8f acc, int mbase, int nbase,
                                          int lh, int hi2,
                                          const float* __restrict__ bias,
                                          char* __restrict__ OutB,
                                          int ldo, int colOff, int mode,
                                          const float* __restrict__ pos)
{
    int col = nbase + lh;
    float bv = bias ? bias[colOff + col] : 0.f;
#pragma unroll
    for (int r = 0; r < 8; ++r) {
        int row = mbase + hi2 * 8 + r;
        float val = acc[r] + bv;
        if (mode == 0) {
            ((bf16*)OutB)[(size_t)row * ldo + colOff + col] = (bf16)val;
        } else if (mode == 1) {
            int bb = row / SVAL, s = row - bb * SVAL;
            ((bf16*)OutB)[((size_t)bb * SPAD + s) * 256 + colOff + col] = (bf16)val;
        } else if (mode == 2) {
            int bb = row / SVAL, rem = row - bb * SVAL;
            int f = rem / PX, p = rem - f * PX;
            int wq = p / 7, hq = p - wq * 7;
            int c = colOff + col;
            int flat = ((wq * 8 + f) * 7 + hq) * 256 + c;
            int cc = flat / SVAL, mm = flat - cc * SVAL;
            ((bf16*)OutB)[((size_t)bb * SPAD + mm) * 256 + cc] = (bf16)val;
        } else if (mode == 3) {
            ((float*)OutB)[(size_t)row * ldo + col] = val;
        } else { // mode 4
            int bb = row / SVAL, s = row - bb * SVAL;
            int f = s / PX, p = s - f * PX;
            val += pos[(size_t)f * (256 * PX) + (size_t)col * PX + p];
            ((bf16*)OutB)[((size_t)bb * 256 + col) * SPAD + s] = (bf16)val;
        }
    }
}

// ---------------------------------------------------------------------------
// Generic bf16 WMMA GEMM: Out[m][n] = sum_k A[m][k] * Wm[n][k] (+bias[n])
// Wave tile = 32x32 (2x2 v_wmma_f32_16x16x32_bf16), 8 waves / 256-thread block.
// Batched via blockIdx.y with element strides As/Ws and byte stride OsBytes.
// Requires M%32==0, N%32==0, K%32==0, EXEC all-ones (uniform control flow).
// ---------------------------------------------------------------------------
__global__ __launch_bounds__(256)
void gemm_wmma_kernel(const bf16* __restrict__ A, int lda,
                      const bf16* __restrict__ Wm, int ldw,
                      const float* __restrict__ bias,
                      void* __restrict__ Out, int ldo, int colOff, int mode,
                      int Mt, int Nt, int K,
                      const float* __restrict__ pos,
                      long As, long Ws, long OsBytes)
{
    int wave = threadIdx.x >> 5;
    int lane = threadIdx.x & 31;
    int wt = blockIdx.x * 8 + wave;
    if (wt >= Mt * Nt) return;                 // wave-uniform exit
    int tm = wt / Nt, tn = wt - tm * Nt;

    A  += (size_t)blockIdx.y * As;
    Wm += (size_t)blockIdx.y * Ws;
    char* OutB = (char*)Out + (size_t)blockIdx.y * OsBytes;

    int lh  = lane & 15;
    int hi2 = lane >> 4;
    int m0 = tm * 32, n0 = tn * 32;

    const bf16* a0p = A  + (size_t)(m0 + lh) * lda + hi2 * 8;
    const bf16* a1p = a0p + (size_t)16 * lda;
    const bf16* b0p = Wm + (size_t)(n0 + lh) * ldw + hi2 * 8;
    const bf16* b1p = b0p + (size_t)16 * ldw;

    v8f c00 = {0.f,0.f,0.f,0.f,0.f,0.f,0.f,0.f};
    v8f c01 = {0.f,0.f,0.f,0.f,0.f,0.f,0.f,0.f};
    v8f c10 = {0.f,0.f,0.f,0.f,0.f,0.f,0.f,0.f};
    v8f c11 = {0.f,0.f,0.f,0.f,0.f,0.f,0.f,0.f};

    for (int k0 = 0; k0 < K; k0 += 32) {
        if (k0 + 128 < K) {   // look-ahead prefetch (global_prefetch_b8)
            __builtin_prefetch((const void*)(a0p + k0 + 128), 0, 1);
            __builtin_prefetch((const void*)(b0p + k0 + 128), 0, 1);
        }
        v16bf a0 = ld_afrag(a0p + k0);
        v16bf a1 = ld_afrag(a1p + k0);
        v16bf b0 = ld_bfrag(b0p + k0);
        v16bf b1 = ld_bfrag(b1p + k0);
        c00 = __builtin_amdgcn_wmma_f32_16x16x32_bf16(false, a0, false, b0, (short)0, c00, false, false);
        c01 = __builtin_amdgcn_wmma_f32_16x16x32_bf16(false, a0, false, b1, (short)0, c01, false, false);
        c10 = __builtin_amdgcn_wmma_f32_16x16x32_bf16(false, a1, false, b0, (short)0, c10, false, false);
        c11 = __builtin_amdgcn_wmma_f32_16x16x32_bf16(false, a1, false, b1, (short)0, c11, false, false);
    }

    store_acc(c00, m0,      n0,      lh, hi2, bias, OutB, ldo, colOff, mode, pos);
    store_acc(c01, m0,      n0 + 16, lh, hi2, bias, OutB, ldo, colOff, mode, pos);
    store_acc(c10, m0 + 16, n0,      lh, hi2, bias, OutB, ldo, colOff, mode, pos);
    store_acc(c11, m0 + 16, n0 + 16, lh, hi2, bias, OutB, ldo, colOff, mode, pos);
}

// ---------------------------------------------------------------------------
// Helpers: f32->bf16 convert, channel-major -> token-major transpose, zero fill
// ---------------------------------------------------------------------------
__global__ void cvt_kernel(const float* __restrict__ src, bf16* __restrict__ dst, int n) {
    int i = blockIdx.x * blockDim.x + threadIdx.x;
    if (i < n) dst[i] = (bf16)src[i];
}

__global__ void tokenize_kernel(const float* __restrict__ src, bf16* __restrict__ dst, int total) {
    int i = blockIdx.x * blockDim.x + threadIdx.x;
    if (i >= total) return;
    int p = i % PX;
    int c = (i / PX) % CIN_;
    int n = i / (PX * CIN_);
    dst[((size_t)(n * PX + p)) * CIN_ + c] = (bf16)src[i];
}

__global__ void zero_kernel(unsigned int* __restrict__ p, long n) {
    long i = (long)blockIdx.x * blockDim.x + threadIdx.x;
    if (i < n) p[i] = 0u;
}

// ---------------------------------------------------------------------------
// Row softmax over 392 valid columns of a 416-wide row; writes bf16 probs with
// zero padding (rows >= 392 fully zeroed). One 416-thread block (13 waves) / row.
// ---------------------------------------------------------------------------
__global__ __launch_bounds__(416)
void softmax_kernel(const float* __restrict__ Sc, bf16* __restrict__ P) {
    __shared__ float wred[16];
    __shared__ float srow;
    int row = blockIdx.x;               // b*416 + n
    int n = row % SPAD;
    int t = threadIdx.x;
    const float* src = Sc + (size_t)row * SPAD;
    bf16* dst = P + (size_t)row * SPAD;
    if (n >= SVAL) { if (t < SPAD) dst[t] = (bf16)0.f; return; }

    float v = (t < SVAL) ? src[t] : -3.402823e38f;
    float m = v;
#pragma unroll
    for (int o = 16; o > 0; o >>= 1) m = fmaxf(m, __shfl_xor(m, o, 32));
    if ((t & 31) == 0) wred[t >> 5] = m;
    __syncthreads();
    if (t < 32) {
        float z = (t < 13) ? wred[t] : -3.402823e38f;
#pragma unroll
        for (int o = 16; o > 0; o >>= 1) z = fmaxf(z, __shfl_xor(z, o, 32));
        if (t == 0) srow = z;
    }
    __syncthreads();
    float mx = srow;
    float e = (t < SVAL) ? __expf(v - mx) : 0.f;
    float s = e;
#pragma unroll
    for (int o = 16; o > 0; o >>= 1) s += __shfl_xor(s, o, 32);
    __syncthreads();
    if ((t & 31) == 0) wred[t >> 5] = s;
    __syncthreads();
    if (t < 32) {
        float z = (t < 13) ? wred[t] : 0.f;
#pragma unroll
        for (int o = 16; o > 0; o >>= 1) z += __shfl_xor(z, o, 32);
        if (t == 0) srow = z;
    }
    __syncthreads();
    float inv = 1.f / srow;
    if (t < SPAD) dst[t] = (bf16)((t < SVAL) ? e * inv : 0.f);
}

// ---------------------------------------------------------------------------
// Final residual: out = (gamma*y + x)*y + x, y from padded f32 Ybuf[b*416+s][c]
// ---------------------------------------------------------------------------
__global__ void final_kernel(const float* __restrict__ x, const float* __restrict__ Y,
                             const float* __restrict__ gamma, float* __restrict__ out,
                             int total) {
    int i = blockIdx.x * blockDim.x + threadIdx.x;
    if (i >= total) return;
    int p = i % PX;
    int c = (i / PX) % CIN_;
    int n = i / (PX * CIN_);
    int bb = n / 8, f = n - bb * 8;
    int s = f * PX + p;
    float y = Y[((size_t)bb * SPAD + s) * CIN_ + c];
    float xv = x[i];
    float g = gamma[0];
    out[i] = (g * y + xv) * y + xv;
}

// ---------------------------------------------------------------------------
// Host-side orchestration
// ---------------------------------------------------------------------------
extern "C" void kernel_launch(void* const* d_in, const int* in_sizes, int n_in,
                              void* d_out, int out_size, void* d_ws, size_t ws_size,
                              hipStream_t stream) {
    (void)in_sizes; (void)n_in; (void)out_size; (void)ws_size;

    const float* x        = (const float*)d_in[0];
    const float* domainX  = (const float*)d_in[1];
    // d_in[2] domainY, d_in[3] w_pre, d_in[4] b_pre: dead computation, skipped
    const float* w_conv3  = (const float*)d_in[5];
    const float* b_conv3  = (const float*)d_in[6];
    const float* w_value  = (const float*)d_in[7];
    const float* b_value  = (const float*)d_in[8];
    const float* b_l1down   = (const float*)d_in[9];
    const float* b_l1up     = (const float*)d_in[10];
    const float* b_l1down_Y = (const float*)d_in[11];
    const float* b_l1up_Y   = (const float*)d_in[12];
    const float* w_l2down   = (const float*)d_in[13];
    const float* b_l2down   = (const float*)d_in[14];
    const float* w_l2up     = (const float*)d_in[15];
    const float* b_l2up     = (const float*)d_in[16];
    const float* w_l2down_Y = (const float*)d_in[17];
    const float* b_l2down_Y = (const float*)d_in[18];
    const float* w_l2up_Y   = (const float*)d_in[19];
    const float* b_l2up_Y   = (const float*)d_in[20];
    const float* w_c11 = (const float*)d_in[21];
    const float* b_c11 = (const float*)d_in[22];
    const float* w_c12 = (const float*)d_in[23];
    const float* b_c12 = (const float*)d_in[24];
    const float* w_c21 = (const float*)d_in[25];
    const float* b_c21 = (const float*)d_in[26];
    const float* w_c22 = (const float*)d_in[27];
    const float* b_c22 = (const float*)d_in[28];
    const float* pos   = (const float*)d_in[29];
    const float* gamma = (const float*)d_in[30];

    // Workspace carve
    char* base = (char*)d_ws;
    size_t off = 0;
    auto carve = [&](size_t bytes) -> char* {
        char* p = base + off;
        off = (off + bytes + 255) & ~(size_t)255;
        return p;
    };
    bf16* Xt  = (bf16*)carve((size_t)NTOK * CIN_ * 2);
    bf16* Dt  = (bf16*)carve((size_t)NTOK * CIN_ * 2);
    bf16* Wc3 = (bf16*)carve((size_t)CMID * CIN_ * 2);
    bf16* Wvl = (bf16*)carve((size_t)CIN_ * CMID * 2);
    bf16* W2d = (bf16*)carve((size_t)CMID * CIN_ * 2);
    bf16* W2u = (bf16*)carve((size_t)CIN_ * CMID * 2);
    bf16* W2dY = (bf16*)carve((size_t)CMID * CIN_ * 2);
    bf16* W2uY = (bf16*)carve((size_t)CIN_ * CMID * 2);
    bf16* Wq1 = (bf16*)carve((size_t)CH_ * CIN_ * 2);
    bf16* Wq2 = (bf16*)carve((size_t)CH_ * CIN_ * 2);
    bf16* Wk1 = (bf16*)carve((size_t)CH_ * CIN_ * 2);
    bf16* Wk2 = (bf16*)carve((size_t)CH_ * CIN_ * 2);
    bf16* tb256  = (bf16*)carve((size_t)NTOK * CMID * 2);
    bf16* tb2048 = (bf16*)carve((size_t)NTOK * CIN_ * 2);
    bf16* dXY    = (bf16*)carve((size_t)NTOK * CIN_ * 2);
    bf16* Qp = (bf16*)carve((size_t)NBATCH * SPAD * 256 * 2);
    bf16* Kt = (bf16*)carve((size_t)NBATCH * SPAD * 256 * 2);
    bf16* Vt = (bf16*)carve((size_t)NBATCH * 256 * SPAD * 2);
    float* Sc = (float*)carve((size_t)NBATCH * SPAD * SPAD * 4);
    bf16* Pm  = (bf16*)carve((size_t)NBATCH * SPAD * SPAD * 2);
    bf16* Ob  = (bf16*)carve((size_t)NBATCH * SPAD * 256 * 2);
    float* Yb = (float*)carve((size_t)NBATCH * SPAD * CIN_ * 4);

    auto cvt = [&](const float* s, bf16* d, int n) {
        cvt_kernel<<<(n + 255) / 256, 256, 0, stream>>>(s, d, n);
    };
    auto gemm = [&](const bf16* A, int lda, const bf16* Wm, int ldw, const float* bias,
                    void* Outp, int ldo, int colOff, int mode,
                    int M, int N, int K, const float* posp,
                    int batches, long As, long Ws, long OsBytes) {
        int Mt = M / 32, Nt = N / 32;
        int tiles = Mt * Nt;
        dim3 g((tiles + 7) / 8, batches);
        gemm_wmma_kernel<<<g, 256, 0, stream>>>(A, lda, Wm, ldw, bias, Outp, ldo,
                                                colOff, mode, Mt, Nt, K, posp,
                                                As, Ws, OsBytes);
    };

    // 1) weights -> bf16
    cvt(w_conv3,   Wc3,  CMID * CIN_);
    cvt(w_value,   Wvl,  CIN_ * CMID);
    cvt(w_l2down,  W2d,  CMID * CIN_);
    cvt(w_l2up,    W2u,  CIN_ * CMID);
    cvt(w_l2down_Y, W2dY, CMID * CIN_);
    cvt(w_l2up_Y,   W2uY, CIN_ * CMID);
    cvt(w_c11, Wq1, CH_ * CIN_);
    cvt(w_c12, Wq2, CH_ * CIN_);
    cvt(w_c21, Wk1, CH_ * CIN_);
    cvt(w_c22, Wk2, CH_ * CIN_);

    // 2) activations -> token-major bf16
    const int totalX = N_IMG * CIN_ * PX;
    tokenize_kernel<<<(totalX + 255) / 256, 256, 0, stream>>>(x, Xt, totalX);
    tokenize_kernel<<<(totalX + 255) / 256, 256, 0, stream>>>(domainX, Dt, totalX);

    // 3) zero padded attention operand buffers
    long zc = (long)NBATCH * SPAD * 256 * 2 / 4;
    zero_kernel<<<(int)((zc + 255) / 256), 256, 0, stream>>>((unsigned int*)Qp, zc);
    zero_kernel<<<(int)((zc + 255) / 256), 256, 0, stream>>>((unsigned int*)Kt, zc);
    zero_kernel<<<(int)((zc + 255) / 256), 256, 0, stream>>>((unsigned int*)Vt, zc);

    // 4) dY chain (shared layer1 weights, _Y biases) then q-part-2 projection
    gemm(Dt, CIN_, Wc3, CIN_, b_l1down_Y, tb256, CMID, 0, 0, NTOK, CMID, CIN_, nullptr, 1, 0, 0, 0);
    gemm(tb256, CMID, Wvl, CMID, b_l1up_Y, tb2048, CIN_, 0, 0, NTOK, CIN_, CMID, nullptr, 1, 0, 0, 0);
    gemm(tb2048, CIN_, W2dY, CIN_, b_l2down_Y, tb256, CMID, 0, 0, NTOK, CMID, CIN_, nullptr, 1, 0, 0, 0);
    gemm(tb256, CMID, W2uY, CMID, b_l2up_Y, dXY, CIN_, 0, 0, NTOK, CIN_, CMID, nullptr, 1, 0, 0, 0);
    gemm(dXY, CIN_, Wq2, CIN_, b_c12, Qp, 0, CH_, 1, NTOK, CH_, CIN_, nullptr, 1, 0, 0, 0);

    // 5) dX chain then k-part-2 projection (odd-permute store)
    gemm(Dt, CIN_, Wc3, CIN_, b_l1down, tb256, CMID, 0, 0, NTOK, CMID, CIN_, nullptr, 1, 0, 0, 0);
    gemm(tb256, CMID, Wvl, CMID, b_l1up, tb2048, CIN_, 0, 0, NTOK, CIN_, CMID, nullptr, 1, 0, 0, 0);
    gemm(tb2048, CIN_, W2d, CIN_, b_l2down, tb256, CMID, 0, 0, NTOK, CMID, CIN_, nullptr, 1, 0, 0, 0);
    gemm(tb256, CMID, W2u, CMID, b_l2up, dXY, CIN_, 0, 0, NTOK, CIN_, CMID, nullptr, 1, 0, 0, 0);
    gemm(dXY, CIN_, Wk2, CIN_, b_c22, Kt, 0, CH_, 2, NTOK, CH_, CIN_, nullptr, 1, 0, 0, 0);

    // 6) q/k part-1 projections from x
    gemm(Xt, CIN_, Wq1, CIN_, b_c11, Qp, 0, 0, 1, NTOK, CH_, CIN_, nullptr, 1, 0, 0, 0);
    gemm(Xt, CIN_, Wk1, CIN_, b_c21, Kt, 0, 0, 2, NTOK, CH_, CIN_, nullptr, 1, 0, 0, 0);

    // 7) value conv (+pos_embed), stored transposed (C, S) per batch
    gemm(Xt, CIN_, Wc3, CIN_, b_conv3, Vt, 0, 0, 4, NTOK, CMID, CIN_, pos, 1, 0, 0, 0);

    // 8) attention scores: Sc[b] = Qp[b] (416x256) x Kt[b]^T-layout (416x256 rows)
    gemm(Qp, 256, Kt, 256, nullptr, Sc, SPAD, 0, 3, SPAD, SPAD, 256, nullptr,
         NBATCH, (long)SPAD * 256, (long)SPAD * 256, (long)SPAD * SPAD * 4);

    // 9) softmax over 392 valid columns
    softmax_kernel<<<NBATCH * SPAD, SPAD, 0, stream>>>(Sc, Pm);

    // 10) out = P (416x416) x Vt^T-layout (256x416 rows) -> Ob[b] (416x256) bf16
    gemm(Pm, SPAD, Vt, SPAD, nullptr, Ob, 256, 0, 0, SPAD, 256, SPAD, nullptr,
         NBATCH, (long)SPAD * SPAD, (long)256 * SPAD, (long)SPAD * 256 * 2);

    // 11) up-projection with w_value: Yb (1664x2048) f32
    gemm(Ob, 256, Wvl, 256, b_value, Yb, CIN_, 0, 3, NBATCH * SPAD, CIN_, 256,
         nullptr, 1, 0, 0, 0);

    // 12) final residual
    final_kernel<<<(totalX + 255) / 256, 256, 0, stream>>>(x, Yb, gamma,
                                                           (float*)d_out, totalX);
}
